// SelfAttention2d_65249143161263
// MI455X (gfx1250) — matrix operand order repaired and checked by reading, not compiled, not run on hardware
//
#include <hip/hip_runtime.h>
#include <hip/hip_bf16.h>
#include <stdint.h>

// ---------------------------------------------------------------------------
// SelfAttention2d on gfx1250 (wave32).  GN1 -> QKV wmma-GEMM -> flash
// attention (S^T = K*Q^T formulation; P^T built in registers via shfl_xor)
// -> out-proj wmma-GEMM -> GN2 + residual.
// All matmuls: v_wmma_f32_16x16x32_bf16.  All tile staging: b128 both sides.
// Flash K/V tiles are double-buffered to hide global-load latency.
// ---------------------------------------------------------------------------

typedef __attribute__((ext_vector_type(16))) __bf16 v16bf;
typedef __attribute__((ext_vector_type(8)))  float  v8f;

union FragAB {
  v16bf v;
  uint4 u[2];
  unsigned int w[8];
};

__device__ __forceinline__ unsigned short f2bf(float f) {
  unsigned int u = __float_as_uint(f);
  u += 0x7FFFu + ((u >> 16) & 1u);   // round-to-nearest-even
  return (unsigned short)(u >> 16);
}
__device__ __forceinline__ unsigned int pack_bf2(float lo, float hi) {
#if __has_builtin(__builtin_amdgcn_cvt_pk_bf16_f32)
  auto r = __builtin_amdgcn_cvt_pk_bf16_f32(lo, hi);
  return __builtin_bit_cast(unsigned int, r);
#else
  return (unsigned int)f2bf(lo) | ((unsigned int)f2bf(hi) << 16);
#endif
}

static constexpr int CH = 128;     // channels
static constexpr int HW = 4096;    // 64*64 spatial
static constexpr int NB = 2;       // batch
static constexpr int NH = 4;       // heads
static constexpr int DH = 32;      // head dim
static constexpr float ATT_SCALE = 0.42044820762685725f;  // 32^-0.25

// ---------------------------------------------------------------------------
__device__ __forceinline__ void block_reduce2(float& s, float& s2) {
  __shared__ float red[2][8];
  for (int off = 1; off < 32; off <<= 1) {
    s  += __shfl_xor(s,  off, 32);
    s2 += __shfl_xor(s2, off, 32);
  }
  const int wid = threadIdx.x >> 5;
  if ((threadIdx.x & 31) == 0) { red[0][wid] = s; red[1][wid] = s2; }
  __syncthreads();
  float ts = 0.f, ts2 = 0.f;
  for (int j = 0; j < 8; ++j) { ts += red[0][j]; ts2 += red[1][j]; }
  s = ts; s2 = ts2;
}

// ---------------------------------------------------------------------------
// Kernel 0: one-shot weight convert + transpose to bf16 [d][k] so both GEMMs
// stage B tiles with pure uint4 copies (no per-block fp32->bf16 conversion).
// ---------------------------------------------------------------------------
__global__ void __launch_bounds__(256)
wcvt_kernel(const float* __restrict__ w_qkv, const float* __restrict__ w_out,
            unsigned short* __restrict__ wqT, unsigned short* __restrict__ woT) {
  const int idx = blockIdx.x * 256 + threadIdx.x;
  if (idx < 384 * 128) {
    const int d = idx >> 7, c = idx & 127;
    wqT[idx] = f2bf(w_qkv[c * 384 + d]);
  }
  const int j = idx - 384 * 128;
  if (j >= 0 && j < 128 * 128) {
    const int d = j >> 7, c = j & 127;
    woT[j] = f2bf(w_out[c * 128 + d]);
  }
}

// ---------------------------------------------------------------------------
// Kernel 1: GroupNorm1.  Writes xn fp32 [n][c][s] (residual) and a bf16 copy
// transposed to [n][s][c] so GEMM A-staging is contiguous.
// ---------------------------------------------------------------------------
__global__ void __launch_bounds__(256)
gn1_kernel(const float* __restrict__ x,
           const float* __restrict__ g1s, const float* __restrict__ g1o,
           float* __restrict__ xn, unsigned short* __restrict__ xnbT) {
  const int ng = blockIdx.x;               // n*16 + g
  const int n = ng >> 4, g = ng & 15;
  const size_t base = (size_t)(n * CH + g * 8) * HW;
  float s = 0.f, s2 = 0.f;
  for (int i = threadIdx.x; i < 8 * HW; i += 256) {
    float v = x[base + i];
    s += v; s2 += v * v;
  }
  block_reduce2(s, s2);
  const float inv  = 1.0f / (8.0f * HW);
  const float mean = s * inv;
  const float var  = s2 * inv - mean * mean;
  const float rstd = rsqrtf(var + 1e-5f);
  for (int i = threadIdx.x; i < 8 * HW; i += 256) {
    const int c = g * 8 + (i >> 12);
    const int sp = i & (HW - 1);
    const float v = (x[base + i] - mean) * rstd * g1s[c] + g1o[c];
    xn[base + i] = v;
    xnbT[(size_t)(n * HW + sp) * CH + c] = f2bf(v);
  }
}

// ---------------------------------------------------------------------------
// Kernel 2: QKV projection.  Q/K written [n][h][s][32] (scaled by 32^-0.25);
// V written transposed [n][h][32][s] (single packed uint4 store per subtile).
// ---------------------------------------------------------------------------
__global__ void __launch_bounds__(128)
qkv_gemm_kernel(const unsigned short* __restrict__ xnbT,  // [n][4096][128] bf16
                const unsigned short* __restrict__ wqT,   // [384][128] bf16
                const float* __restrict__ b_qkv,          // [384]
                unsigned short* __restrict__ qb,
                unsigned short* __restrict__ kb,
                unsigned short* __restrict__ vTb) {
  __shared__ __align__(16) unsigned short lds_a[64 * 40];  // [s][k]
  __shared__ __align__(16) unsigned short lds_b[64 * 40];  // [d][k]
  const int s0 = blockIdx.x * 64;
  const int d0 = blockIdx.y * 64;
  const int n  = blockIdx.z;
  const int tid = threadIdx.x, lane = tid & 31, wid = tid >> 5;
  const int srw = (wid >> 1) * 32, dcw = (wid & 1) * 32;
  const int l15 = lane & 15, hi8 = (lane >> 4) * 8;

  v8f acc[2][2] = {};
  for (int kk0 = 0; kk0 < CH; kk0 += 32) {
    for (int i = 0; i < 2; ++i) {                    // A tile: uint4 both sides
      const int lin = tid + i * 128;                 // 0..255
      const int s = lin >> 2, k8 = lin & 3;
      *(uint4*)&lds_a[s * 40 + k8 * 8] =
          *(const uint4*)&xnbT[(size_t)(n * HW + s0 + s) * CH + kk0 + k8 * 8];
    }
    for (int i = 0; i < 2; ++i) {                    // B tile: uint4 both sides
      const int lin = tid + i * 128;
      const int d = lin >> 2, k8 = lin & 3;
      *(uint4*)&lds_b[d * 40 + k8 * 8] =
          *(const uint4*)&wqT[(size_t)(d0 + d) * CH + kk0 + k8 * 8];
    }
    __syncthreads();
    FragAB afr[2], bfr[2];
    for (int r = 0; r < 2; ++r) {
      const unsigned short* p =
          &lds_a[(srw + r * 16 + l15) * 40 + ((lane < 16) ? 0 : 8)];
      afr[r].u[0] = *(const uint4*)(p);
      afr[r].u[1] = *(const uint4*)(p + 16);
    }
    for (int cc = 0; cc < 2; ++cc) {
      const unsigned short* p =
          &lds_b[(dcw + cc * 16 + l15) * 40 + ((lane < 16) ? 0 : 8)];
      bfr[cc].u[0] = *(const uint4*)(p);
      bfr[cc].u[1] = *(const uint4*)(p + 16);
    }
    for (int r = 0; r < 2; ++r)
      for (int cc = 0; cc < 2; ++cc)
        acc[r][cc] = __builtin_amdgcn_wmma_f32_16x16x32_bf16(
            false, afr[r].v, false, bfr[cc].v, (short)0, acc[r][cc], false, false);
    __syncthreads();
  }
  for (int r = 0; r < 2; ++r)
    for (int cc = 0; cc < 2; ++cc) {
      const int dbase = d0 + dcw + cc * 16;
      const int sel = dbase >> 7;               // 0=q 1=k 2=v (tile-uniform)
      const int d = dbase + l15;
      const int hd = d & 127;
      const int head = hd >> 5, dh = hd & 31;
      const int sbase = s0 + srw + r * 16 + hi8;
      const float bias = b_qkv[d];
      const v8f& a = acc[r][cc];
      if (sel == 2) {        // V: 8 consecutive s -> one packed uint4 store
        uint4 pk;
        pk.x = pack_bf2(a[0] + bias, a[1] + bias);
        pk.y = pack_bf2(a[2] + bias, a[3] + bias);
        pk.z = pack_bf2(a[4] + bias, a[5] + bias);
        pk.w = pack_bf2(a[6] + bias, a[7] + bias);
        *(uint4*)&vTb[((size_t)(n * NH + head) * DH + dh) * HW + sbase] = pk;
      } else {               // Q/K: stride-DH scalar stores, pointer bump
        unsigned short* p = (sel == 0 ? qb : kb) +
            ((size_t)(n * NH + head) * HW + sbase) * DH + dh;
        for (int vv = 0; vv < 8; ++vv)
          p[vv * DH] = f2bf((a[vv] + bias) * ATT_SCALE);
      }
    }
}

// ---------------------------------------------------------------------------
// Kernel 3: flash attention, transposed-S formulation, double-buffered tiles.
//   S^T = K * Q^T   (A = K tile from LDS, B = Q straight from global)
//   O^T = V^T * P^T (A = V^T tile from LDS, B = P^T built by shfl_xor(16))
// Block = 4 waves x 16 queries (Br=64); key tiles Bc=64, 64 iterations.
// ---------------------------------------------------------------------------
__global__ void __launch_bounds__(128)
flash_attn_kernel(const unsigned short* __restrict__ q,   // [nh][s][32]
                  const unsigned short* __restrict__ k,   // [nh][s][32]
                  const unsigned short* __restrict__ vT,  // [nh][32][s]
                  unsigned short* __restrict__ yt) {      // [n][s][128]
  __shared__ __align__(16) unsigned short kt[2][64 * 40]; // [key][dh]
  __shared__ __align__(16) unsigned short vt[2][32 * 72]; // [dh][key]
  const int nh = blockIdx.y;                               // n*4 + head
  const int tid = threadIdx.x, lane = tid & 31, wid = tid >> 5;
  const int l15 = lane & 15;
  const int q0w = blockIdx.x * 64 + wid * 16;              // this wave's queries
  const size_t base = (size_t)nh * HW * DH;
  const unsigned short* qp  = q  + base;
  const unsigned short* kp  = k  + base;
  const unsigned short* vtp = vT + base;

  // per-thread staging coordinates (two b128 chunks each for K and V^T)
  const int kkey0 = tid >> 2,         kc80 = (tid & 3) * 8;
  const int kkey1 = (tid + 128) >> 2, kc81 = kc80;          // second chunk
  const int vdh0 = tid >> 3,          vk80 = (tid & 7) * 8;
  const int vdh1 = (tid + 128) >> 3,  vk81 = vk80;

  uint4 kr0, kr1, vr0, vr1;
  auto load_regs = [&](int kb0) {
    kr0 = *(const uint4*)&kp[(size_t)(kb0 + kkey0) * DH + kc80];
    kr1 = *(const uint4*)&kp[(size_t)(kb0 + kkey1) * DH + kc81];
    vr0 = *(const uint4*)&vtp[(size_t)vdh0 * HW + kb0 + vk80];
    vr1 = *(const uint4*)&vtp[(size_t)vdh1 * HW + kb0 + vk81];
  };
  auto store_regs = [&](int b) {
    *(uint4*)&kt[b][kkey0 * 40 + kc80] = kr0;
    *(uint4*)&kt[b][kkey1 * 40 + kc81] = kr1;
    *(uint4*)&vt[b][vdh0 * 72 + vk80] = vr0;
    *(uint4*)&vt[b][vdh1 * 72 + vk81] = vr1;
  };

  // Q as B-fragment: lane column = query l15; low lanes dh 0-7,16-23, high 8-15,24-31.
  FragAB qf;
  {
    const unsigned short* p = qp + (size_t)(q0w + l15) * DH + ((lane < 16) ? 0 : 8);
    qf.u[0] = *(const uint4*)(p);
    qf.u[1] = *(const uint4*)(p + 16);
  }
  v8f o0 = {}, o1 = {};
  float m_run = -1e30f, l_run = 0.f;

  load_regs(0);
  store_regs(0);
  __syncthreads();

  constexpr int NT = HW / 64;
  for (int t = 0; t < NT; ++t) {
    const int cur = t & 1;
    if (t + 1 < NT) load_regs((t + 1) * 64);   // issue next tile loads now

    // S^T subtiles: rows = keys (16 per subtile), cols = queries.
    v8f sa[4];
    for (int c = 0; c < 4; ++c) {
      FragAB af;
      const unsigned short* p =
          &kt[cur][(c * 16 + l15) * 40 + ((lane < 16) ? 0 : 8)];
      af.u[0] = *(const uint4*)(p);
      af.u[1] = *(const uint4*)(p + 16);
      v8f z = {};
      sa[c] = __builtin_amdgcn_wmma_f32_16x16x32_bf16(false, af.v, false, qf.v,
                                                      (short)0, z, false, false);
    }
    // softmax over keys (rows): per-lane scalar stats for query l15.
    float tmax = -1e30f;
    for (int c = 0; c < 4; ++c)
      for (int vv = 0; vv < 8; ++vv) tmax = fmaxf(tmax, sa[c][vv]);
    tmax = fmaxf(tmax, __shfl_xor(tmax, 16, 32));
    const float mn = fmaxf(m_run, tmax);
    const float corr = __expf(m_run - mn);
    m_run = mn;

    unsigned int pack[4][4];
    float ps = 0.f;
    for (int c = 0; c < 4; ++c)
      for (int j = 0; j < 4; ++j) {
        const float a = __expf(sa[c][2 * j]     - mn);
        const float b = __expf(sa[c][2 * j + 1] - mn);
        ps += a + b;
        pack[c][j] = pack_bf2(a, b);
      }
    ps += __shfl_xor(ps, 16, 32);
    l_run = l_run * corr + ps;
    for (int vv = 0; vv < 8; ++vv) { o0[vv] *= corr; o1[vv] *= corr; }

    // O^T += V^T * P^T : P^T B-fragment built from registers + shfl_xor(16).
    for (int ch = 0; ch < 2; ++ch) {
      const int cl = 2 * ch, cu = 2 * ch + 1;
      FragAB pf;
      for (int j = 0; j < 4; ++j) {
        const unsigned int swl = (unsigned int)__shfl_xor((int)pack[cl][j], 16, 32);
        const unsigned int swu = (unsigned int)__shfl_xor((int)pack[cu][j], 16, 32);
        pf.w[j]     = (lane < 16) ? pack[cl][j] : swu;
        pf.w[4 + j] = (lane < 16) ? swl : pack[cu][j];
      }
      for (int t2 = 0; t2 < 2; ++t2) {
        FragAB vf;                                  // A rows = dh t2*16+l15
        const unsigned short* pv =
            &vt[cur][(t2 * 16 + l15) * 72 + ch * 32 + ((lane < 16) ? 0 : 16)];
        vf.u[0] = *(const uint4*)(pv);
        vf.u[1] = *(const uint4*)(pv + 8);
        v8f& oo = t2 ? o1 : o0;
        oo = __builtin_amdgcn_wmma_f32_16x16x32_bf16(false, vf.v, false, pf.v,
                                                     (short)0, oo, false, false);
      }
    }
    if (t + 1 < NT) store_regs(1 - cur);       // waits loads, fills other buf
    __syncthreads();
  }
  // epilogue: lane holds query s = q0w+l15, dh = t2*16 + hi8 + v (consecutive)
  const int n = nh >> 2, head = nh & 3;
  const int hi8 = (lane >> 4) * 8;
  const float inv_l = 1.0f / l_run;
  const int s = q0w + l15;
  for (int t2 = 0; t2 < 2; ++t2) {
    v8f& oo = t2 ? o1 : o0;
    uint4 pk;
    pk.x = pack_bf2(oo[0] * inv_l, oo[1] * inv_l);
    pk.y = pack_bf2(oo[2] * inv_l, oo[3] * inv_l);
    pk.z = pack_bf2(oo[4] * inv_l, oo[5] * inv_l);
    pk.w = pack_bf2(oo[6] * inv_l, oo[7] * inv_l);
    *(uint4*)&yt[(size_t)(n * HW + s) * CH + head * DH + t2 * 16 + hi8] = pk;
  }
}

// ---------------------------------------------------------------------------
// Kernel 4: out-projection.  att_out[row,d] = sum_c yt[row,c]*w_out[c,d]+b.
// ---------------------------------------------------------------------------
__global__ void __launch_bounds__(128)
out_gemm_kernel(const unsigned short* __restrict__ yt,    // [8192][128] bf16
                const unsigned short* __restrict__ woT,   // [128][128] bf16 [d][k]
                const float* __restrict__ b_out,          // [128]
                float* __restrict__ att_out) {            // [8192][128] fp32
  __shared__ __align__(16) unsigned short lds_a[64 * 40];
  __shared__ __align__(16) unsigned short lds_b[64 * 40];
  const int r0 = blockIdx.x * 64;
  const int d0 = blockIdx.y * 64;
  const int tid = threadIdx.x, lane = tid & 31, wid = tid >> 5;
  const int srw = (wid >> 1) * 32, dcw = (wid & 1) * 32;
  const int l15 = lane & 15, hi8 = (lane >> 4) * 8;

  v8f acc[2][2] = {};
  for (int kk0 = 0; kk0 < CH; kk0 += 32) {
    for (int i = 0; i < 2; ++i) {
      const int lin = tid + i * 128;
      const int s = lin >> 2, k8 = lin & 3;
      *(uint4*)&lds_a[s * 40 + k8 * 8] =
          *(const uint4*)&yt[(size_t)(r0 + s) * CH + kk0 + k8 * 8];
    }
    for (int i = 0; i < 2; ++i) {
      const int lin = tid + i * 128;
      const int d = lin >> 2, k8 = lin & 3;
      *(uint4*)&lds_b[d * 40 + k8 * 8] =
          *(const uint4*)&woT[(size_t)(d0 + d) * CH + kk0 + k8 * 8];
    }
    __syncthreads();
    FragAB afr[2], bfr[2];
    for (int r = 0; r < 2; ++r) {
      const unsigned short* p =
          &lds_a[(srw + r * 16 + l15) * 40 + ((lane < 16) ? 0 : 8)];
      afr[r].u[0] = *(const uint4*)(p);
      afr[r].u[1] = *(const uint4*)(p + 16);
    }
    for (int cc = 0; cc < 2; ++cc) {
      const unsigned short* p =
          &lds_b[(dcw + cc * 16 + l15) * 40 + ((lane < 16) ? 0 : 8)];
      bfr[cc].u[0] = *(const uint4*)(p);
      bfr[cc].u[1] = *(const uint4*)(p + 16);
    }
    for (int r = 0; r < 2; ++r)
      for (int cc = 0; cc < 2; ++cc)
        acc[r][cc] = __builtin_amdgcn_wmma_f32_16x16x32_bf16(
            false, afr[r].v, false, bfr[cc].v, (short)0, acc[r][cc], false, false);
    __syncthreads();
  }
  for (int r = 0; r < 2; ++r)
    for (int cc = 0; cc < 2; ++cc) {
      const int d = d0 + dcw + cc * 16 + l15;
      const float bias = b_out[d];
      float* p = &att_out[(size_t)(r0 + srw + r * 16 + hi8) * CH + d];
      for (int vv = 0; vv < 8; ++vv)
        p[vv * CH] = acc[r][cc][vv] + bias;
    }
}

// ---------------------------------------------------------------------------
// Kernel 5: GroupNorm2 statistics over att_out [n][s][c].
// ---------------------------------------------------------------------------
__global__ void __launch_bounds__(256)
gn2_stats_kernel(const float* __restrict__ att, float* __restrict__ stats) {
  const int ng = blockIdx.x;
  const int n = ng >> 4, g = ng & 15;
  const float* b = att + (size_t)n * HW * CH + g * 8;
  float s = 0.f, s2 = 0.f;
  for (int i = threadIdx.x; i < HW * 8; i += 256) {
    const int srow = i >> 3, j = i & 7;
    const float vv = b[(size_t)srow * CH + j];
    s += vv; s2 += vv * vv;
  }
  block_reduce2(s, s2);
  if (threadIdx.x == 0) {
    const float inv  = 1.0f / (8.0f * HW);
    const float mean = s * inv;
    const float var  = s2 * inv - mean * mean;
    stats[ng * 2]     = mean;
    stats[ng * 2 + 1] = rsqrtf(var + 1e-5f);
  }
}

// ---------------------------------------------------------------------------
// Kernel 6: out = xn + GN2(att_out), final layout [n][c][h][w].
// ---------------------------------------------------------------------------
__global__ void __launch_bounds__(256)
final_kernel(const float* __restrict__ xn, const float* __restrict__ att,
             const float* __restrict__ stats,
             const float* __restrict__ g2s, const float* __restrict__ g2o,
             float* __restrict__ out) {
  const int idx = blockIdx.x * 256 + threadIdx.x;
  if (idx >= NB * CH * HW) return;
  const int n = idx >> 19;
  const int rem = idx & (CH * HW - 1);
  const int c = rem >> 12;
  const int s = rem & (HW - 1);
  const int g = c >> 3;
  const float mean = stats[(n * 16 + g) * 2];
  const float rstd = stats[(n * 16 + g) * 2 + 1];
  const float o = att[((size_t)n * HW + s) * CH + c];
  out[idx] = xn[idx] + (o - mean) * rstd * g2s[c] + g2o[c];
}

// ---------------------------------------------------------------------------
extern "C" void kernel_launch(void* const* d_in, const int* in_sizes, int n_in,
                              void* d_out, int out_size, void* d_ws, size_t ws_size,
                              hipStream_t stream) {
  (void)in_sizes; (void)n_in; (void)out_size; (void)ws_size;
  const float* x     = (const float*)d_in[0];
  const float* w_qkv = (const float*)d_in[1];
  const float* b_qkv = (const float*)d_in[2];
  const float* w_out = (const float*)d_in[3];
  const float* b_out = (const float*)d_in[4];
  const float* g1s   = (const float*)d_in[5];
  const float* g1o   = (const float*)d_in[6];
  const float* g2s   = (const float*)d_in[7];
  const float* g2o   = (const float*)d_in[8];
  float* out = (float*)d_out;

  char* ws = (char*)d_ws;
  const size_t MB = 1024 * 1024;
  float*          xn      = (float*)(ws + 0);                 // 4 MB
  float*          att_out = (float*)(ws + 4 * MB);            // 4 MB
  unsigned short* xnbT    = (unsigned short*)(ws + 8 * MB);   // 2 MB
  unsigned short* qb      = (unsigned short*)(ws + 10 * MB);  // 2 MB
  unsigned short* kb      = (unsigned short*)(ws + 12 * MB);  // 2 MB
  unsigned short* vTb     = (unsigned short*)(ws + 14 * MB);  // 2 MB
  unsigned short* yt      = (unsigned short*)(ws + 16 * MB);  // 2 MB
  float*          stats   = (float*)(ws + 18 * MB);           // 256 B
  unsigned short* wqT     = (unsigned short*)(ws + 18 * MB + 4096);   // 96 KB
  unsigned short* woT     = (unsigned short*)(ws + 18 * MB + 4096 + 384 * 128 * 2);

  wcvt_kernel<<<256, 256, 0, stream>>>(w_qkv, w_out, wqT, woT);
  gn1_kernel<<<32, 256, 0, stream>>>(x, g1s, g1o, xn, xnbT);
  qkv_gemm_kernel<<<dim3(64, 6, 2), 128, 0, stream>>>(xnbT, wqT, b_qkv, qb, kb, vTb);
  flash_attn_kernel<<<dim3(64, 8), 128, 0, stream>>>(qb, kb, vTb, yt);
  out_gemm_kernel<<<dim3(128, 2), 128, 0, stream>>>(yt, woT, b_out, att_out);
  gn2_stats_kernel<<<32, 256, 0, stream>>>(att_out, stats);
  final_kernel<<<4096, 256, 0, stream>>>(xn, att_out, stats, g2s, g2o, out);
}
